// BERT_69226282877621
// MI455X (gfx1250) — matrix operand order repaired and checked by reading, not compiled, not run on hardware
//
#include <hip/hip_runtime.h>
#include <hip/hip_bf16.h>

// ---------------------------------------------------------------------------
// BERT-base forward on gfx1250 (MI455X), wave32 + WMMA f32_16x16x32_f16.
// Compute-bound (~1.55 TFLOP vs ~340MB weight traffic @ 23.3TB/s): all
// matmuls on the matrix pipes, activations resident in f16 (fits in 192MB
// L2), fp32 accumulation + fp32 residual/LayerNorm path.
//
// - Weights are converted fp32->f16 ONCE per use into a pre-swizzled tiled
//   image whose memory layout equals the LDS fragment layout, so GEMM
//   B-staging is a pure linear async global->LDS copy (no VALU, no strides).
// - LDS tiles in *WMMA fragment order* ([tile][kChunk][lane][16 halves]):
//   every fragment is one 32B LDS read (2x ds_load_b128).
// - GEMM is 2-stage double-buffered; A and B tiles staged with gfx1250
//   async global->LDS b128 copies (ASYNCcnt + s_wait_asynccnt).
// - Dependent-WMMA chains broken with rotating accumulators (f16 WMMA->WMMA
//   RAW hazard = ~5 slots per the ISA hazard table).
// ---------------------------------------------------------------------------

typedef __attribute__((ext_vector_type(16))) _Float16 v16h;
typedef __attribute__((ext_vector_type(8)))  float    v8f;
typedef __attribute__((__vector_size__(16))) int      v4i;

#define DD   768
#define LL   512
#define NLAY 12
#define HH   12
#define DKK  64
#define BB   16
#define FFD  3072
#define MM   (BB * LL)   // 8192 token rows

#define TM 64
#define TN 128
#define TK 64
#define TILE_ELEMS (TK * TN)   // 8192 halves = 16 KB per (k0,n0) tile

#if defined(__gfx1250__) && __has_builtin(__builtin_amdgcn_global_load_async_to_lds_b128)
#define HAS_ASYNC_LDS 1
typedef __attribute__((address_space(1))) v4i* gptr_b128;
typedef __attribute__((address_space(3))) v4i* lptr_b128;
#define ASYNC_CP16(gsrc, ldst)                                               \
  __builtin_amdgcn_global_load_async_to_lds_b128((gptr_b128)(gsrc),          \
                                                 (lptr_b128)(ldst), 0, 0)
#else
#define HAS_ASYNC_LDS 0
#define ASYNC_CP16(gsrc, ldst)                                               \
  (*reinterpret_cast<uint4*>(ldst) = *reinterpret_cast<const uint4*>(gsrc))
#endif

// ===========================================================================
// Embedding: h[b,l,:] = token_emb[x[b,l],:] + pe[l,:]   (fp32 + f16 copy)
// ===========================================================================
__global__ __launch_bounds__(256)
void embed_kernel(const int* __restrict__ x, const float* __restrict__ tok,
                  const float* __restrict__ pe, float* __restrict__ h,
                  _Float16* __restrict__ hf) {
  const int row = blockIdx.x;        // b*L + l
  const int l   = row & (LL - 1);
  const int t   = x[row];
#pragma unroll
  for (int i = 0; i < 3; ++i) {
    const int c = threadIdx.x + i * 256;
    const float v = tok[(size_t)t * DD + c] + pe[(size_t)l * DD + c];
    h[(size_t)row * DD + c]  = v;
    hf[(size_t)row * DD + c] = (_Float16)v;
  }
}

// ===========================================================================
// Weight pre-swizzle: W (fp32, KxN row-major) -> Wh (f16, tiled fragment
// order). Tile (k0,n0) occupies TILE_ELEMS contiguous halves laid out as
// [nt(8)][kc(2)][lane(32)][j(16)] — identical to the GEMM's LDS image, so
// the GEMM can stage it with a flat async copy. Reads are coalesced across
// the wave (consecutive lanes -> consecutive n).
// ===========================================================================
__global__ __launch_bounds__(256)
void convert_w_kernel(const float* __restrict__ W, _Float16* __restrict__ Wh,
                      int Ndim, int Kdim) {
  const int tid = threadIdx.x;
  const int k0  = blockIdx.x * TK;
  const int n0  = blockIdx.y * TN;
  const int ktiles = Kdim / TK;
  const size_t tbase = ((size_t)blockIdx.y * ktiles + blockIdx.x) * TILE_ELEMS;

  const int bn  = tid & 127;          // n within tile
  const int bkg = (tid >> 7) * 32;    // 0 or 32
  const int bnt = bn >> 4;
  const int bkc = bkg >> 5;
  const int bln = bn & 15;

  const float* wp = &W[(size_t)(k0 + bkg) * Ndim + n0 + bn];
  v16h lo, hi;
#pragma unroll
  for (int k = 0; k < 16; ++k) lo[k] = (_Float16)wp[(size_t)k * Ndim];
#pragma unroll
  for (int k = 0; k < 16; ++k) hi[k] = (_Float16)wp[(size_t)(16 + k) * Ndim];

  const size_t base = tbase + (size_t)((bnt * 2 + bkc) * 32) * 16;
  *reinterpret_cast<v16h*>(&Wh[base + (size_t)bln * 16])        = lo;
  *reinterpret_cast<v16h*>(&Wh[base + (size_t)(bln + 16) * 16]) = hi;
}

// ===========================================================================
// WMMA GEMM:  O = A(f16, MxK) * Wh(f16 pre-swizzled) + bias.
// Block tile 64x128, K-step 64, double-buffered async staging. 256 thr =
// 8 waves; wave(wm 0..3, wn 0..1) owns 16x64 = 4 accumulators, 8 WMMA/stage.
// EPI: 0 = bias->f16, 1 = bias->f32, 2 = bias+GELU->f16,
//      3 = bias->f16 scattered into V^T layout [b][h][d][kp].
// ===========================================================================
template <int EPI>
__global__ __launch_bounds__(256)
void gemm_wmma(const _Float16* __restrict__ A, const _Float16* __restrict__ Wh,
               const float* __restrict__ bias, float* __restrict__ Of32,
               _Float16* __restrict__ Of16, int Mdim, int Ndim, int Kdim) {
  __shared__ __align__(32) _Float16 As[2][4][2][32][16];   // 16 KB
  __shared__ __align__(32) _Float16 Bs[2][8][2][32][16];   // 32 KB

  const int tid  = threadIdx.x;
  const int lane = tid & 31;
  const int wave = tid >> 5;
  const int wm   = wave >> 1;
  const int wn   = wave & 1;
  const int m0   = blockIdx.x * TM;
  const int n0   = blockIdx.y * TN;
  const int ktiles = Kdim / TK;

  v8f acc[4];
#pragma unroll
  for (int t = 0; t < 4; ++t)
#pragma unroll
    for (int i = 0; i < 8; ++i) acc[t][i] = 0.0f;

  // A stager: thread -> (row 0..63, 16-half k-group)
  const int arow = tid >> 2;
  const int akg  = (tid & 3) * 16;          // 0,16,32,48
  const int akc  = akg >> 5;                // k-chunk 0/1
  const int ajb  = (akg & 16) >> 1;         // j base 0/8

  auto stageA = [&](int k0s, int buf) {
    const _Float16* src = &A[(size_t)(m0 + arow) * Kdim + k0s + akg];
    ASYNC_CP16(src,     &As[buf][arow >> 4][akc][(arow & 15) + (akg & 16)][0]);
    ASYNC_CP16(src + 8, &As[buf][arow >> 4][akc][(arow & 15) + (akg & 16)][8]);
  };
  // B stager: Wh tile layout == LDS layout -> flat 64B/thread copy
  auto stageB = [&](int k0s, int buf) {
    const _Float16* src =
        &Wh[((size_t)blockIdx.y * ktiles + (k0s / TK)) * TILE_ELEMS + tid * 32];
    _Float16* dst = &Bs[buf][0][0][0][0] + tid * 32;
#pragma unroll
    for (int i = 0; i < 4; ++i) ASYNC_CP16(src + i * 8, dst + i * 8);
  };

  // ---- prologue: stage tile 0 ----------------------------------------
  stageA(0, 0);
  stageB(0, 0);
#if HAS_ASYNC_LDS
  asm volatile("s_wait_asynccnt 0" ::: "memory");
#endif
  __syncthreads();

  // ---- main pipelined loop -------------------------------------------
  for (int k0 = 0; k0 < Kdim; k0 += TK) {
    const int  buf  = (k0 / TK) & 1;
    const int  nxt  = buf ^ 1;
    const bool more = (k0 + TK) < Kdim;

    if (more) {                 // copies for tile i+1 fly across the WMMAs
      stageA(k0 + TK, nxt);
      stageB(k0 + TK, nxt);
    }

#pragma unroll
    for (int kc = 0; kc < 2; ++kc) {
      const v16h a = *reinterpret_cast<const v16h*>(&As[buf][wm][kc][lane][0]);
#pragma unroll
      for (int t = 0; t < 4; ++t) {
        const v16h b =
            *reinterpret_cast<const v16h*>(&Bs[buf][wn * 4 + t][kc][lane][0]);
        acc[t] = __builtin_amdgcn_wmma_f32_16x16x32_f16(
            false, a, false, b, (short)0, acc[t], false, false);
      }
    }

#if HAS_ASYNC_LDS
    if (more) asm volatile("s_wait_asynccnt 0" ::: "memory");
#endif
    __syncthreads();
  }

  // ---- epilogue: C layout -> VGPR r is row (r | 8+r by lane-half) -----
  const int rbase = (lane & 16) ? 8 : 0;
  const int ncol  = lane & 15;
#pragma unroll
  for (int t = 0; t < 4; ++t) {
    const int gc = n0 + wn * 64 + t * 16 + ncol;
    const float bv = bias[gc];
#pragma unroll
    for (int r = 0; r < 8; ++r) {
      const int gr = m0 + wm * 16 + rbase + r;
      float xv = acc[t][r] + bv;
      if (EPI == 1) {
        Of32[(size_t)gr * Ndim + gc] = xv;
      } else if (EPI == 3) {
        // V^T layout: [b][h][d][kp]
        const int bbx = gr >> 9;          // / LL
        const int kp  = gr & (LL - 1);
        const int hx  = gc >> 6;          // / DKK
        const int dx  = gc & 63;
        Of16[(((size_t)bbx * HH + hx) * DKK + dx) * LL + kp] = (_Float16)xv;
      } else {
        if (EPI == 2) xv = 0.5f * xv * (1.0f + erff(xv * 0.70710678118f));
        Of16[(size_t)gr * Ndim + gc] = (_Float16)xv;
      }
    }
  }
}

// ===========================================================================
// Attention: one block = (b, head, 16-row q tile). 128 threads = 4 waves.
// scores(16x512) = q k^T (WMMA, split accumulators -> no RAW chain) ->
// mask -> softmax (unnormalized f16 probs in fragment order) ->
// ctx(16x64) = p V (WMMA, 4 rotating accumulators), row-sum normalize.
// ===========================================================================
__global__ __launch_bounds__(128)
void attn_wmma(const _Float16* __restrict__ Q, const _Float16* __restrict__ Kf,
               const _Float16* __restrict__ Vt, const int* __restrict__ x,
               _Float16* __restrict__ ctx) {
  __shared__ __align__(32) _Float16 qs[2][32][16];       // q frags (2 k-chunks)
  __shared__ float    sc[16][LL];                        // 32 KB scores
  __shared__ __align__(32) _Float16 wf[16][32][16];      // prob frags, 16 ksteps
  __shared__ float    rbuf[16][8];
  __shared__ float    rmax[16];
  __shared__ float    rsum[16];

  const int tid  = threadIdx.x;
  const int lane = tid & 31;
  const int wave = tid >> 5;       // 0..3
  const int qt = blockIdx.x;       // 0..31
  const int hh = blockIdx.y;       // 0..11
  const int bb = blockIdx.z;       // 0..15
  const int q0 = qt * 16;

  // load q tile 16x64 directly into fragment order
  {
    const int r = tid >> 3;               // 0..15
    const int c = (tid & 7) * 8;          // 0..56
    const int chunk = c >> 5;
    const int kk = c & 31;
    const int ln = r + ((kk & 16) ? 16 : 0);
    const int jb = kk & 8;                // 0 or 8
    const uint4 v = *reinterpret_cast<const uint4*>(
        &Q[((size_t)(bb * LL + q0 + r)) * DD + hh * DKK + c]);
    *reinterpret_cast<uint4*>(&qs[chunk][ln][jb]) = v;
  }
  __syncthreads();

  const int ncol  = lane & 15;
  const int kbB   = (lane & 16) ? 16 : 0;
  const int rbase = (lane & 16) ? 8 : 0;

  const v16h a0 = *reinterpret_cast<const v16h*>(&qs[0][lane][0]);
  const v16h a1 = *reinterpret_cast<const v16h*>(&qs[1][lane][0]);

  const float scale = 0.125f;  // 1/sqrt(64)
  for (int kt = wave; kt < 32; kt += 4) {
    const int kp = kt * 16 + ncol;
    const size_t kbase = ((size_t)(bb * LL + kp)) * DD + hh * DKK;
    const v16h b0 = *reinterpret_cast<const v16h*>(&Kf[kbase + 0  + kbB]);
    const v16h b1 = *reinterpret_cast<const v16h*>(&Kf[kbase + 32 + kbB]);
    v8f ae, ao;
#pragma unroll
    for (int i = 0; i < 8; ++i) { ae[i] = 0.0f; ao[i] = 0.0f; }
    // independent accumulators: no WMMA->WMMA RAW hazard
    ae = __builtin_amdgcn_wmma_f32_16x16x32_f16(false, a0, false, b0, (short)0, ae, false, false);
    ao = __builtin_amdgcn_wmma_f32_16x16x32_f16(false, a1, false, b1, (short)0, ao, false, false);
    const v8f s = ae + ao;
    const float mval = (x[bb * LL + kp] > 0) ? 0.0f : -1e9f;
#pragma unroll
    for (int r = 0; r < 8; ++r)
      sc[rbase + r][kp] = s[r] * scale + mval;
  }
  __syncthreads();

  // softmax: 8 threads per row scan 64 cols each; probs stored frag-ordered
  {
    const int row = tid >> 3;
    const int sub = tid & 7;
    float mx = -3.4e38f;
    for (int i = 0; i < 64; ++i) mx = fmaxf(mx, sc[row][sub * 64 + i]);
    rbuf[row][sub] = mx;
    __syncthreads();
    if (sub == 0) {
      float m = rbuf[row][0];
#pragma unroll
      for (int i = 1; i < 8; ++i) m = fmaxf(m, rbuf[row][i]);
      rmax[row] = m;
    }
    __syncthreads();
    const float m = rmax[row];
    float ps = 0.0f;
    for (int i = 0; i < 64; ++i) {
      const int col = sub * 64 + i;
      const float e = __expf(sc[row][col] - m);
      wf[col >> 5][row + ((col & 16) ? 16 : 0)][col & 15] =
          (_Float16)e;
      ps += e;
    }
    rbuf[row][sub] = ps;
    __syncthreads();
    if (sub == 0) {
      float s = 0.0f;
#pragma unroll
      for (int i = 0; i < 8; ++i) s += rbuf[row][i];
      rsum[row] = s;
    }
  }
  __syncthreads();

  // ctx: wave owns d-tile `wave`; 4 rotating accumulators (RAW distance 4)
  v8f cacc[4];
#pragma unroll
  for (int t = 0; t < 4; ++t)
#pragma unroll
    for (int i = 0; i < 8; ++i) cacc[t][i] = 0.0f;
  const int dcol = wave * 16 + ncol;
  const size_t vrow = (((size_t)bb * HH + hh) * DKK + dcol) * LL;
#pragma unroll 4
  for (int ks = 0; ks < 16; ++ks) {
    const v16h a = *reinterpret_cast<const v16h*>(&wf[ks][lane][0]);
    const v16h b = *reinterpret_cast<const v16h*>(&Vt[vrow + ks * 32 + kbB]);
    cacc[ks & 3] = __builtin_amdgcn_wmma_f32_16x16x32_f16(
        false, a, false, b, (short)0, cacc[ks & 3], false, false);
  }
  const v8f ctot = (cacc[0] + cacc[1]) + (cacc[2] + cacc[3]);
#pragma unroll
  for (int r = 0; r < 8; ++r) {
    const int row = rbase + r;
    const float vv = ctot[r] / rsum[row];
    ctx[((size_t)(bb * LL + q0 + row)) * DD + hh * DKK + dcol] = (_Float16)vv;
  }
}

// ===========================================================================
// Fused residual-add + LayerNorm (gamma/beta shared across both norm sites
// per layer, matching the reference). One block per token row.
// ===========================================================================
__global__ __launch_bounds__(256)
void add_ln_kernel(const float* __restrict__ src, const float* __restrict__ res,
                   const float* __restrict__ g, const float* __restrict__ b,
                   float* __restrict__ outf, _Float16* __restrict__ outh) {
  __shared__ float red[256];
  const int row = blockIdx.x;
  const int tid = threadIdx.x;
  float v[3];
  float s = 0.0f;
#pragma unroll
  for (int i = 0; i < 3; ++i) {
    const int c = tid + i * 256;
    v[i] = src[(size_t)row * DD + c] + res[(size_t)row * DD + c];
    s += v[i];
  }
  red[tid] = s;
  __syncthreads();
  for (int off = 128; off > 0; off >>= 1) {
    if (tid < off) red[tid] += red[tid + off];
    __syncthreads();
  }
  const float mu = red[0] * (1.0f / DD);
  __syncthreads();
  float s2 = 0.0f;
#pragma unroll
  for (int i = 0; i < 3; ++i) { const float d = v[i] - mu; s2 += d * d; }
  red[tid] = s2;
  __syncthreads();
  for (int off = 128; off > 0; off >>= 1) {
    if (tid < off) red[tid] += red[tid + off];
    __syncthreads();
  }
  const float rs = rsqrtf(red[0] * (1.0f / DD) + 1e-5f);
#pragma unroll
  for (int i = 0; i < 3; ++i) {
    const int c = tid + i * 256;
    const float o = (v[i] - mu) * rs * g[c] + b[c];
    outf[(size_t)row * DD + c] = o;
    outh[(size_t)row * DD + c] = (_Float16)o;
  }
}

// ===========================================================================
extern "C" void kernel_launch(void* const* d_in, const int* in_sizes, int n_in,
                              void* d_out, int out_size, void* d_ws, size_t ws_size,
                              hipStream_t stream) {
  const int*   x    = (const int*)d_in[0];
  const float* tok  = (const float*)d_in[1];
  const float* pe   = (const float*)d_in[2];
  const float* Wq   = (const float*)d_in[3];
  const float* bq   = (const float*)d_in[4];
  const float* Wk   = (const float*)d_in[5];
  const float* bk   = (const float*)d_in[6];
  const float* Wv   = (const float*)d_in[7];
  const float* bv   = (const float*)d_in[8];
  const float* Wo   = (const float*)d_in[9];
  const float* bo   = (const float*)d_in[10];
  const float* W1   = (const float*)d_in[11];
  const float* b1   = (const float*)d_in[12];
  const float* W2   = (const float*)d_in[13];
  const float* b2   = (const float*)d_in[14];
  const float* ln_g = (const float*)d_in[15];
  const float* ln_b = (const float*)d_in[16];

  // workspace carve-out
  char* w = (char*)d_ws;
  float*    h    = (float*)w;    w += (size_t)MM * DD * 4;
  float*    tmp  = (float*)w;    w += (size_t)MM * DD * 4;
  _Float16* hf   = (_Float16*)w; w += (size_t)MM * DD * 2;
  _Float16* qf   = (_Float16*)w; w += (size_t)MM * DD * 2;
  _Float16* kf   = (_Float16*)w; w += (size_t)MM * DD * 2;
  _Float16* vt   = (_Float16*)w; w += (size_t)MM * DD * 2;   // V^T [b][h][d][kp]
  _Float16* ctxf = (_Float16*)w; w += (size_t)MM * DD * 2;
  _Float16* uf   = (_Float16*)w; w += (size_t)MM * FFD * 2;
  _Float16* wh   = (_Float16*)w; w += (size_t)DD * FFD * 2;  // swizzled weights

  embed_kernel<<<MM, 256, 0, stream>>>(x, tok, pe, h, hf);

  const dim3 gProj(MM / TM, DD / TN);     // 128 x 6
  const dim3 gFF1 (MM / TM, FFD / TN);    // 128 x 24
  const dim3 gAttn(LL / 16, HH, BB);      // 32 x 12 x 16
  const dim3 cDxD (DD / TK, DD / TN);     // 12 x 6
  const dim3 cDxF (DD / TK, FFD / TN);    // 12 x 24
  const dim3 cFxD (FFD / TK, DD / TN);    // 48 x 6

  for (int l = 0; l < NLAY; ++l) {
    const float* wq = Wq + (size_t)l * DD * DD;
    const float* wk = Wk + (size_t)l * DD * DD;
    const float* wv = Wv + (size_t)l * DD * DD;
    const float* wo = Wo + (size_t)l * DD * DD;
    const float* w1 = W1 + (size_t)l * DD * FFD;
    const float* w2 = W2 + (size_t)l * FFD * DD;

    convert_w_kernel<<<cDxD, 256, 0, stream>>>(wq, wh, DD, DD);
    gemm_wmma<0><<<gProj, 256, 0, stream>>>(hf, wh, bq + (size_t)l * DD, nullptr, qf, MM, DD, DD);
    convert_w_kernel<<<cDxD, 256, 0, stream>>>(wk, wh, DD, DD);
    gemm_wmma<0><<<gProj, 256, 0, stream>>>(hf, wh, bk + (size_t)l * DD, nullptr, kf, MM, DD, DD);
    convert_w_kernel<<<cDxD, 256, 0, stream>>>(wv, wh, DD, DD);
    gemm_wmma<3><<<gProj, 256, 0, stream>>>(hf, wh, bv + (size_t)l * DD, nullptr, vt, MM, DD, DD);

    attn_wmma<<<gAttn, 128, 0, stream>>>(qf, kf, vt, x, ctxf);

    convert_w_kernel<<<cDxD, 256, 0, stream>>>(wo, wh, DD, DD);
    gemm_wmma<1><<<gProj, 256, 0, stream>>>(ctxf, wh, bo + (size_t)l * DD, tmp, nullptr, MM, DD, DD);
    add_ln_kernel<<<MM, 256, 0, stream>>>(tmp, h, ln_g + (size_t)l * DD, ln_b + (size_t)l * DD, h, hf);

    convert_w_kernel<<<cDxF, 256, 0, stream>>>(w1, wh, FFD, DD);
    gemm_wmma<2><<<gFF1, 256, 0, stream>>>(hf, wh, b1 + (size_t)l * FFD, nullptr, uf, MM, FFD, DD);
    convert_w_kernel<<<cFxD, 256, 0, stream>>>(w2, wh, DD, FFD);
    gemm_wmma<1><<<gProj, 256, 0, stream>>>(uf, wh, b2 + (size_t)l * DD, tmp, nullptr, MM, DD, FFD);
    add_ln_kernel<<<MM, 256, 0, stream>>>(tmp, h, ln_g + (size_t)l * DD, ln_b + (size_t)l * DD, h, hf);
  }

  (void)hipMemcpyAsync(d_out, h, (size_t)MM * DD * sizeof(float),
                       hipMemcpyDeviceToDevice, stream);
}
